// Speller_50826642981470
// MI455X (gfx1250) — matrix operand order repaired and checked by reading, not compiled, not run on hardware
//
#include <hip/hip_runtime.h>
#include <hip/hip_bf16.h>
#include <stdint.h>

// ---------------------------------------------------------------------------
// Types for CDNA5 WMMA f16: A/B = v16h (16 halves / lane), C/D = v8f.
// ---------------------------------------------------------------------------
typedef __attribute__((ext_vector_type(16))) _Float16 v16h;
typedef __attribute__((ext_vector_type(8)))  float    v8f;
typedef __attribute__((__vector_size__(16))) int      v4i;

union Frag {
    v16h      v;
    _Float16  h[16];
    uint4     q[2];
};

#define Bsz 128
#define Lsz 512
#define Dsz 512
#define Tsz 200
#define Esz 384
#define Csz 128
#define Hsz 512
#define Vsz 46

// CDNA5 async global->LDS path (ASYNCcnt-tracked), guarded for toolchain
#if defined(__has_builtin)
#if __has_builtin(__builtin_amdgcn_global_load_async_to_lds_b128)
#define USE_ASYNC_LDS 1
#endif
#if __has_builtin(__builtin_amdgcn_s_wait_asynccnt)
#define HAVE_WAIT_ASYNC_BUILTIN 1
#endif
#endif

typedef __attribute__((address_space(1))) v4i gv4i;
typedef __attribute__((address_space(3))) v4i lv4i;

__device__ __forceinline__ void wait_async0() {
#if defined(USE_ASYNC_LDS)
#if defined(HAVE_WAIT_ASYNC_BUILTIN)
    __builtin_amdgcn_s_wait_asynccnt(0);
#else
    asm volatile("s_wait_asynccnt 0" ::: "memory");
#endif
#endif
}

// ---------------------------------------------------------------------------
// Weight pack: Wcat[r, 0:512] = w_ih[r], Wcat[r, 512:1024] = w_hh[r]  (f16)
// bias[r] = b_ih[r] + b_hh[r]
// ---------------------------------------------------------------------------
__global__ __launch_bounds__(256) void pack_w_kernel(
    const float* __restrict__ wih, const float* __restrict__ whh,
    const float* __restrict__ bih, const float* __restrict__ bhh,
    _Float16* __restrict__ Wcat, float* __restrict__ bias)
{
    int idx = blockIdx.x * 256 + threadIdx.x;
    if (idx < 2048 * 1024) {
        int r = idx >> 10, c = idx & 1023;
        float v = (c < 512) ? wih[r * 512 + c] : whh[r * 512 + (c - 512)];
        Wcat[idx] = (_Float16)v;
    }
    if (idx < 2048) bias[idx] = bih[idx] + bhh[idx];
}

__global__ __launch_bounds__(256) void conv_kvw_kernel(
    const float* __restrict__ kw, const float* __restrict__ vw,
    _Float16* __restrict__ Kw, _Float16* __restrict__ Vw)
{
    int i = blockIdx.x * 256 + threadIdx.x;
    if (i < Csz * Dsz) { Kw[i] = (_Float16)kw[i]; Vw[i] = (_Float16)vw[i]; }
}

// ---------------------------------------------------------------------------
// Precompute keyf/valf [B,L,C] (f16) = LF[l,b,:] @ {key,value}_w^T + bias
// WMMA tiles: per wave one (b, 16-l, 16-c) tile, K=512 in 32-chunks.
// ---------------------------------------------------------------------------
__global__ __launch_bounds__(256) void kv_kernel(
    const float* __restrict__ LF,       // [L,B,D]
    const _Float16* __restrict__ Kw,    // [C,D]
    const _Float16* __restrict__ Vw,    // [C,D]
    const float* __restrict__ keyb, const float* __restrict__ valb,
    _Float16* __restrict__ keyf,        // [B,L,C]
    _Float16* __restrict__ valf)
{
    const int lane = threadIdx.x & 31;
    const int wid  = threadIdx.x >> 5;
    const int gw   = blockIdx.x * 8 + wid;         // 0..32767
    const int ct = gw & 7;
    const int lt = (gw >> 3) & 31;
    const int b  = gw >> 8;
    const int l0 = lt * 16, c0 = ct * 16;
    const int half = lane >> 4, lr = lane & 15;

    v8f ak, av;
    {
        float kb = keyb[c0 + lr], vb = valb[c0 + lr];
        #pragma unroll
        for (int e = 0; e < 8; ++e) { ak[e] = kb; av[e] = vb; }
    }

    const float* ar = LF + (size_t)(l0 + lr) * (Bsz * Dsz) + (size_t)b * Dsz;
    for (int kk = 0; kk < Dsz; kk += 32) {
        Frag a;
        const float* p0 = ar + kk + half * 16;
        float4 f0 = *(const float4*)(p0);
        float4 f1 = *(const float4*)(p0 + 4);
        float4 f2 = *(const float4*)(p0 + 8);
        float4 f3 = *(const float4*)(p0 + 12);
        a.h[0] = (_Float16)f0.x; a.h[1] = (_Float16)f0.y; a.h[2] = (_Float16)f0.z; a.h[3] = (_Float16)f0.w;
        a.h[4] = (_Float16)f1.x; a.h[5] = (_Float16)f1.y; a.h[6] = (_Float16)f1.z; a.h[7] = (_Float16)f1.w;
        a.h[8]  = (_Float16)f2.x; a.h[9]  = (_Float16)f2.y; a.h[10] = (_Float16)f2.z; a.h[11] = (_Float16)f2.w;
        a.h[12] = (_Float16)f3.x; a.h[13] = (_Float16)f3.y; a.h[14] = (_Float16)f3.z; a.h[15] = (_Float16)f3.w;

        Frag bk, bv;
        const _Float16* kp = Kw + (c0 + lr) * Dsz + kk + half * 16;
        const _Float16* vp = Vw + (c0 + lr) * Dsz + kk + half * 16;
        bk.q[0] = *(const uint4*)(kp);     bk.q[1] = *(const uint4*)(kp + 8);
        bv.q[0] = *(const uint4*)(vp);     bv.q[1] = *(const uint4*)(vp + 8);

        ak = __builtin_amdgcn_wmma_f32_16x16x32_f16(false, a.v, false, bk.v, (short)0, ak, false, false);
        av = __builtin_amdgcn_wmma_f32_16x16x32_f16(false, a.v, false, bv.v, (short)0, av, false, false);
    }

    #pragma unroll
    for (int e = 0; e < 8; ++e) {
        int l = l0 + e + half * 8;
        int c = c0 + lr;
        size_t o = ((size_t)b * Lsz + l) * Csz + c;
        keyf[o] = (_Float16)ak[e];
        valf[o] = (_Float16)av[e];
    }
}

// ---------------------------------------------------------------------------
// Fused LSTM layer: gates = X[128,1024] @ Wcat^T[1024,2048] + bias, then
// elementwise cell update.
//
// Block = 8 waves = 8 batch tiles (mt=wid), one hidden tile nt=blockIdx.
// Weight k-chunks (64 cols x 32 k = 4KB) are staged once per block into
// double-buffered LDS (async global->LDS on CDNA5 when available) and
// broadcast to all 8 waves via ds_load_b128 -> 8x less weight traffic.
// Each wave holds the four gate tiles (i/f/g/o) sharing one A fragment.
// ---------------------------------------------------------------------------
__global__ __launch_bounds__(256) void lstm_layer_kernel(
    const _Float16* __restrict__ X,     // [128,1024]  (not written here)
    const _Float16* __restrict__ W,     // [2048,1024]
    const float* __restrict__ bias,     // [2048]
    float* __restrict__ cst,            // [128,512] in/out
    float* __restrict__ h_f32,          // [128,512] out, may be null
    _Float16* __restrict__ x_next,      // next layer X (cols 0:512), may be null
    _Float16* __restrict__ hstage)      // [128,512] f16 h for next step
{
    __shared__ _Float16 stage[2][64][32];   // 8 KB double-buffered weight tile

    const int tid  = threadIdx.x;
    const int lane = tid & 31;
    const int wid  = tid >> 5;
    const int nt = blockIdx.x;              // 0..31 hidden tile
    const int mt = wid;                     // 0..7 batch tile
    const int m0 = mt * 16, n0 = nt * 16;
    const int half = lane >> 4, lr = lane & 15;

    // cooperative staging map: thread -> 16B of the 64x32 chunk
    const int sc   = tid >> 2;              // staged col 0..63 (= gate*16 + col)
    const int koff = (tid & 3) * 8;         // 0,8,16,24
    const int g_s  = sc >> 4, lr_s = sc & 15;
    const _Float16* wsrc = W + (size_t)(g_s * 512 + n0 + lr_s) * 1024 + koff;

    auto stage_chunk = [&](int buf, int kk) {
#if defined(USE_ASYNC_LDS)
        __builtin_amdgcn_global_load_async_to_lds_b128(
            (gv4i*)(wsrc + kk),
            (lv4i*)&stage[buf][sc][koff],
            0, 0);
#else
        uint4 t = *(const uint4*)(wsrc + kk);
        *(uint4*)&stage[buf][sc][koff] = t;
#endif
    };

    v8f acc[4];
    #pragma unroll
    for (int g = 0; g < 4; ++g) {
        float bv = bias[g * 512 + n0 + lr];
        #pragma unroll
        for (int e = 0; e < 8; ++e) acc[g][e] = bv;
    }

    stage_chunk(0, 0);

    const _Float16* arow = X + (m0 + lr) * 1024 + half * 16;
    for (int kk = 0, it = 0; kk < 1024; kk += 32, ++it) {
        const int buf = it & 1;
        wait_async0();            // my staged chunk for `buf` has landed
        __syncthreads();          // everyone's chunk visible; prev reads done
        if (kk + 32 < 1024) stage_chunk(buf ^ 1, kk + 32);

        Frag a;
        a.q[0] = *(const uint4*)(arow + kk);
        a.q[1] = *(const uint4*)(arow + kk + 8);
        #pragma unroll
        for (int g = 0; g < 4; ++g) {
            Frag bf;
            const _Float16* bp = &stage[buf][g * 16 + lr][half * 16];
            bf.q[0] = *(const uint4*)(bp);
            bf.q[1] = *(const uint4*)(bp + 8);
            acc[g] = __builtin_amdgcn_wmma_f32_16x16x32_f16(false, a.v, false, bf.v, (short)0, acc[g], false, false);
        }
    }

    const int n = n0 + lr;
    #pragma unroll
    for (int e = 0; e < 8; ++e) {
        int m = m0 + e + half * 8;
        float iv = acc[0][e], fv = acc[1][e], gv = acc[2][e], ov = acc[3][e];
        float c_old = cst[m * 512 + n];
        float ig = 1.f / (1.f + __expf(-iv));
        float fg = 1.f / (1.f + __expf(-fv));
        float og = 1.f / (1.f + __expf(-ov));
        float c2 = fg * c_old + ig * tanhf(gv);
        float h2 = og * tanhf(c2);
        cst[m * 512 + n] = c2;
        if (h_f32) h_f32[m * 512 + n] = h2;
        _Float16 hh = (_Float16)h2;
        if (x_next) x_next[m * 1024 + n] = hh;
        hstage[m * 512 + n] = hh;
    }
}

// ---------------------------------------------------------------------------
// Per-step input pack: X0 embedding slot + copy staged h into X second halves
// ---------------------------------------------------------------------------
__global__ __launch_bounds__(512) void pack_step_kernel(
    const int* __restrict__ transcript, const float* __restrict__ emb,
    const _Float16* __restrict__ st0, const _Float16* __restrict__ st1,
    const _Float16* __restrict__ st2,
    _Float16* __restrict__ X0, _Float16* __restrict__ X1, _Float16* __restrict__ X2,
    int t)
{
    int b = blockIdx.x, tid = threadIdx.x;
    int ch = (t == 0) ? 0 : transcript[(t - 1) * Bsz + b];
    if (tid < Esz) X0[b * 1024 + tid] = (_Float16)emb[ch * Esz + tid];
    X0[b * 1024 + 512 + tid] = st0[b * 512 + tid];
    X1[b * 1024 + 512 + tid] = st1[b * 512 + tid];
    X2[b * 1024 + 512 + tid] = st2[b * 512 + tid];
}

// ---------------------------------------------------------------------------
// Attention: q = h2 @ phi_w^T + phi_b; energy over keyf; masked/renormed
// softmax; ctx over valf. One workgroup per batch element.
// ---------------------------------------------------------------------------
__global__ __launch_bounds__(256) void attend_kernel(
    const float* __restrict__ h2,       // [B,512]
    const _Float16* __restrict__ keyf,  // [B,L,C]
    const _Float16* __restrict__ valf,  // [B,L,C]
    const float* __restrict__ phi_w,    // [C,H]
    const float* __restrict__ phi_b,    // [C]
    const int* __restrict__ lens,       // [B]
    float* __restrict__ attn_out,       // [B,L] slice
    float* __restrict__ ctx_f32,        // [B,C]
    _Float16* __restrict__ X0)          // write f16 ctx at [b,384+c]
{
    int b = blockIdx.x, tid = threadIdx.x;
    __shared__ float sq[Csz];
    __shared__ float se[Lsz];
    __shared__ float red[256];

    if (tid < Csz) {
        float s = phi_b[tid];
        const float* hp = h2 + b * Hsz;
        const float* wp = phi_w + tid * Hsz;
        for (int k = 0; k < Hsz; ++k) s += hp[k] * wp[k];
        sq[tid] = s;
    }
    __syncthreads();

    for (int l = tid; l < Lsz; l += 256) {
        const _Float16* kp = keyf + ((size_t)b * Lsz + l) * Csz;
        float s = 0.f;
        for (int c = 0; c < Csz; ++c) s += sq[c] * (float)kp[c];
        se[l] = s;
    }
    __syncthreads();

    // max
    red[tid] = fmaxf(se[tid], se[tid + 256]);
    __syncthreads();
    for (int o = 128; o > 0; o >>= 1) {
        if (tid < o) red[tid] = fmaxf(red[tid], red[tid + o]);
        __syncthreads();
    }
    float mx = red[0];
    __syncthreads();

    float s0 = __expf(se[tid] - mx);
    float s1 = __expf(se[tid + 256] - mx);
    red[tid] = s0 + s1;
    __syncthreads();
    for (int o = 128; o > 0; o >>= 1) {
        if (tid < o) red[tid] += red[tid + o];
        __syncthreads();
    }
    float sumall = red[0];
    __syncthreads();

    int len = lens[b];
    float p0 = (tid < len) ? s0 : 0.f;
    float p1 = (tid + 256 < len) ? s1 : 0.f;
    red[tid] = p0 + p1;
    __syncthreads();
    for (int o = 128; o > 0; o >>= 1) {
        if (tid < o) red[tid] += red[tid + o];
        __syncthreads();
    }
    float summ = red[0];
    __syncthreads();

    // a = (p/sumall) / max(summ/sumall, 1e-12)
    float inv = 1.f / (sumall * fmaxf(summ / sumall, 1e-12f));
    float a0 = p0 * inv, a1 = p1 * inv;
    se[tid] = a0; se[tid + 256] = a1;
    attn_out[(size_t)b * Lsz + tid]       = a0;
    attn_out[(size_t)b * Lsz + tid + 256] = a1;
    __syncthreads();

    if (tid < Csz) {
        const _Float16* vp = valf + (size_t)b * Lsz * Csz + tid;
        float s = 0.f;
        for (int l = 0; l < Lsz; ++l) s += se[l] * (float)vp[(size_t)l * Csz];
        ctx_f32[b * Csz + tid] = s;
        X0[b * 1024 + Esz + tid] = (_Float16)s;
    }
}

// ---------------------------------------------------------------------------
// Logits: [h2|ctx][B,640] @ out_w^T[640,46] + out_b
// ---------------------------------------------------------------------------
__global__ __launch_bounds__(64) void logits_kernel(
    const float* __restrict__ h2, const float* __restrict__ ctx,
    const float* __restrict__ out_w, const float* __restrict__ out_b,
    float* __restrict__ out)            // [B,V] slice
{
    int b = blockIdx.x, v = threadIdx.x;
    if (v < Vsz) {
        const float* w = out_w + v * (Hsz + Csz);
        float s = out_b[v];
        const float* hp = h2 + b * Hsz;
        for (int k = 0; k < Hsz; ++k) s += hp[k] * w[k];
        const float* cp = ctx + b * Csz;
        for (int k = 0; k < Csz; ++k) s += cp[k] * w[Hsz + k];
        out[b * Vsz + v] = s;
    }
}

// ---------------------------------------------------------------------------
// State init from init_hx*/init_cx* broadcasts
// ---------------------------------------------------------------------------
__global__ __launch_bounds__(256) void init_state_kernel(
    const float* __restrict__ ihx0, const float* __restrict__ icx0,
    const float* __restrict__ ihx1, const float* __restrict__ icx1,
    const float* __restrict__ ihx2, const float* __restrict__ icx2,
    float* __restrict__ c0, float* __restrict__ c1, float* __restrict__ c2,
    _Float16* __restrict__ st0, _Float16* __restrict__ st1, _Float16* __restrict__ st2,
    float* __restrict__ h2f)
{
    int i = blockIdx.x * 256 + threadIdx.x;   // B*H = 65536
    if (i < Bsz * Hsz) {
        int n = i & (Hsz - 1);
        c0[i] = icx0[n]; c1[i] = icx1[n]; c2[i] = icx2[n];
        st0[i] = (_Float16)ihx0[n];
        st1[i] = (_Float16)ihx1[n];
        st2[i] = (_Float16)ihx2[n];
        h2f[i] = ihx2[n];
    }
}

// ---------------------------------------------------------------------------
extern "C" void kernel_launch(void* const* d_in, const int* in_sizes, int n_in,
                              void* d_out, int out_size, void* d_ws, size_t ws_size,
                              hipStream_t stream) {
    const float* LF       = (const float*)d_in[0];
    const int*   lens     = (const int*)d_in[1];
    const int*   transcr  = (const int*)d_in[2];
    const float* emb      = (const float*)d_in[3];
    const float* w_ih[3]  = {(const float*)d_in[4],  (const float*)d_in[8],  (const float*)d_in[12]};
    const float* w_hh[3]  = {(const float*)d_in[5],  (const float*)d_in[9],  (const float*)d_in[13]};
    const float* b_ih[3]  = {(const float*)d_in[6],  (const float*)d_in[10], (const float*)d_in[14]};
    const float* b_hh[3]  = {(const float*)d_in[7],  (const float*)d_in[11], (const float*)d_in[15]};
    const float* phi_w    = (const float*)d_in[16];
    const float* phi_b    = (const float*)d_in[17];
    const float* key_w    = (const float*)d_in[18];
    const float* key_b    = (const float*)d_in[19];
    const float* value_w  = (const float*)d_in[20];
    const float* value_b  = (const float*)d_in[21];
    const float* out_w    = (const float*)d_in[22];
    const float* out_b    = (const float*)d_in[23];
    const float* ihx[3]   = {(const float*)d_in[24], (const float*)d_in[26], (const float*)d_in[28]};
    const float* icx[3]   = {(const float*)d_in[25], (const float*)d_in[27], (const float*)d_in[29]};

    // workspace carve-out (256B aligned)
    size_t off = 0;
    char* base = (char*)d_ws;
    auto carve = [&](size_t bytes) -> void* {
        void* p = base + off;
        off = (off + bytes + 255) & ~(size_t)255;
        return p;
    };
    _Float16* W[3];   for (int l = 0; l < 3; ++l) W[l]   = (_Float16*)carve(2048 * 1024 * sizeof(_Float16));
    float*    bias[3];for (int l = 0; l < 3; ++l) bias[l]= (float*)carve(2048 * sizeof(float));
    _Float16* Kw   = (_Float16*)carve(Csz * Dsz * sizeof(_Float16));
    _Float16* Vw   = (_Float16*)carve(Csz * Dsz * sizeof(_Float16));
    _Float16* keyf = (_Float16*)carve((size_t)Bsz * Lsz * Csz * sizeof(_Float16));
    _Float16* valf = (_Float16*)carve((size_t)Bsz * Lsz * Csz * sizeof(_Float16));
    _Float16* X[3]; for (int l = 0; l < 3; ++l) X[l]  = (_Float16*)carve(Bsz * 1024 * sizeof(_Float16));
    _Float16* st[3];for (int l = 0; l < 3; ++l) st[l] = (_Float16*)carve(Bsz * Hsz * sizeof(_Float16));
    float*    c[3]; for (int l = 0; l < 3; ++l) c[l]  = (float*)carve(Bsz * Hsz * sizeof(float));
    float* h2f  = (float*)carve(Bsz * Hsz * sizeof(float));
    float* ctx  = (float*)carve(Bsz * Csz * sizeof(float));
    float* attn_dummy = (float*)carve((size_t)Bsz * Lsz * sizeof(float));

    float* logits_base = (float*)d_out;                         // [T,B,V]
    float* attn_base   = logits_base + (size_t)Tsz * Bsz * Vsz; // [T,B,L]

    // one-shot setup
    for (int l = 0; l < 3; ++l)
        pack_w_kernel<<<(2048 * 1024 + 255) / 256, 256, 0, stream>>>(
            w_ih[l], w_hh[l], b_ih[l], b_hh[l], W[l], bias[l]);
    conv_kvw_kernel<<<(Csz * Dsz + 255) / 256, 256, 0, stream>>>(key_w, value_w, Kw, Vw);
    kv_kernel<<<4096, 256, 0, stream>>>(LF, Kw, Vw, key_b, value_b, keyf, valf);
    init_state_kernel<<<(Bsz * Hsz + 255) / 256, 256, 0, stream>>>(
        ihx[0], icx[0], ihx[1], icx[1], ihx[2], icx[2],
        c[0], c[1], c[2], st[0], st[1], st[2], h2f);

    // initial context from initial decoder state (attn map discarded)
    attend_kernel<<<Bsz, 256, 0, stream>>>(h2f, keyf, valf, phi_w, phi_b, lens,
                                           attn_dummy, ctx, X[0]);

    for (int t = 0; t < Tsz; ++t) {
        pack_step_kernel<<<Bsz, 512, 0, stream>>>(transcr, emb, st[0], st[1], st[2],
                                                  X[0], X[1], X[2], t);
        lstm_layer_kernel<<<32, 256, 0, stream>>>(X[0], W[0], bias[0], c[0],
                                                  nullptr, X[1], st[0]);
        lstm_layer_kernel<<<32, 256, 0, stream>>>(X[1], W[1], bias[1], c[1],
                                                  nullptr, X[2], st[1]);
        lstm_layer_kernel<<<32, 256, 0, stream>>>(X[2], W[2], bias[2], c[2],
                                                  h2f, nullptr, st[2]);
        attend_kernel<<<Bsz, 256, 0, stream>>>(h2f, keyf, valf, phi_w, phi_b, lens,
                                               attn_base + (size_t)t * Bsz * Lsz, ctx, X[0]);
        logits_kernel<<<Bsz, 64, 0, stream>>>(h2f, ctx, out_w, out_b,
                                              logits_base + (size_t)t * Bsz * Vsz);
    }
}
